// _DCN_Module_8375186227796
// MI455X (gfx1250) — matrix operand order repaired and checked by reading, not compiled, not run on hardware
//
#include <hip/hip_runtime.h>
#include <hip/hip_bf16.h>

typedef __attribute__((ext_vector_type(16))) __bf16 v16bf;
typedef __attribute__((ext_vector_type(8)))  __bf16 v8bf;
typedef __attribute__((ext_vector_type(8)))  float  v8f;

struct BPair { v8bf a, b; };   // 32B, bit_cast to v16bf (no-op in regs)

#define N_SAMPLES 131072
#define DIM       128
#define KCENT     512
#define NT        (KCENT / 16)   // 32 center tiles
#define CPG       64             // centers per scatter group
#define GROUPS    8
#define CHUNKS    32
#define STRH      136            // LDS tile row stride in bf16 elems (16B-aligned, bank-rotating)

// ---- workspace layout (dword offsets) ----
#define WS_C2     0                          // 512 f32 center norms
#define WS_LOSS   512                        // 1 f32 loss accumulator
#define WS_LABELS 513                        // 131072 int labels
#define WS_SUMS   (WS_LABELS + N_SAMPLES)    // 512*128 f32 per-center sums
#define WS_CNT    (WS_SUMS + KCENT*DIM)      // 512 int per-center add counts
#define WS_CH     (WS_CNT + KCENT)           // 512*128 bf16 hi  (32768 dwords)
#define WS_CL     (WS_CH + KCENT*DIM/2)      // 512*128 bf16 lo  (32768 dwords)

__global__ void k_initws(float* __restrict__ sums, int* __restrict__ cnts,
                         float* __restrict__ lossacc) {
    int idx = blockIdx.x * 256 + threadIdx.x;
    if (idx < KCENT * DIM) sums[idx] = 0.0f;
    if (idx < KCENT)       cnts[idx] = 0;
    if (idx == 0)          *lossacc  = 0.0f;
}

// One block per center: f32 -> bf16 hi + bf16 lo (separate contiguous arrays); reduce ||c||^2.
__global__ __launch_bounds__(DIM) void k_prep(const float* __restrict__ centers,
                                              float* __restrict__ c2,
                                              __bf16* __restrict__ ch,
                                              __bf16* __restrict__ cl) {
    __shared__ float red[DIM];
    int c = blockIdx.x, d = threadIdx.x;
    float x  = centers[c * DIM + d];
    __bf16 h = (__bf16)x;
    float lo = x - (float)h;
    ch[c * DIM + d] = h;
    cl[c * DIM + d] = (__bf16)lo;
    red[d] = x * x;
    __syncthreads();
    for (int s = DIM / 2; s > 0; s >>= 1) {
        if (d < s) red[d] += red[d + s];
        __syncthreads();
    }
    if (d == 0) c2[c] = red[0];
}

// 4 waves/block, 32 rows/wave (two 16-row A register tiles). bf16x2-split GEMM:
// 6 independent WMMA accumulator chains, 24 WMMAs per 16 ds_load_b128, double-buffered
// LDS center tiles with a single barrier per tile (staging overlaps the WMMA burst).
__global__ __launch_bounds__(128)
void k_dist_argmin(const float* __restrict__ emb,
                   const __bf16* __restrict__ ch, const __bf16* __restrict__ cl,
                   const float* __restrict__ c2, int* __restrict__ labels,
                   float* __restrict__ labels_f, float* __restrict__ lossacc) {
    __shared__ __align__(16) __bf16 tileH[2][16 * STRH];   // 2 x 4352 B
    __shared__ __align__(16) __bf16 tileL[2][16 * STRH];
    __shared__ float c2s[KCENT];

    const int tid  = threadIdx.x;
    const int lane = tid & 31;
    const int wave = tid >> 5;
    const int half = lane >> 4;               // wave32 half (ISA K-split)
    const int lm   = lane & 15;
    const int rowBase = blockIdx.x * 128 + wave * 32;

    const unsigned* __restrict__ chd = (const unsigned*)ch;   // 2 bf16 / dword
    const unsigned* __restrict__ cld = (const unsigned*)cl;

    auto stage = [&](int ct) {
        unsigned* tH = (unsigned*)tileH[ct & 1];
        unsigned* tL = (unsigned*)tileL[ct & 1];
        for (int j = tid; j < 16 * 64; j += 128) {
            int n = j >> 6, kd = j & 63;
            tH[n * (STRH / 2) + kd] = chd[(ct * 16 + n) * 64 + kd];
            tL[n * (STRH / 2) + kd] = cld[(ct * 16 + n) * 64 + kd];
        }
    };

    for (int j = tid; j < KCENT; j += 128) c2s[j] = c2[j];
    stage(0);

    // ---- A: 2 tiles x 16 rows x 128 f32 -> split bf16 hi/lo (contiguous k = 16*half + i, matches B) ----
    v16bf ah[2][4], al[2][4];
    float rowx2[2][8];
#pragma unroll
    for (int p = 0; p < 2; ++p) {
        const int m = rowBase + p * 16 + lm;   // both halves hold rows 0..15 of the tile
        float x2p = 0.0f;
#pragma unroll
        for (int t = 0; t < 4; ++t) {
#pragma unroll
            for (int i = 0; i < 16; ++i) {
                int k = 32 * t + 16 * half + i;
                float x  = emb[m * DIM + k];
                __bf16 h = (__bf16)x;
                float lo = x - (float)h;
                ah[p][t][i] = h;
                al[p][t][i] = (__bf16)lo;
                x2p += x * x;
            }
        }
        float x2full = x2p + __shfl_xor(x2p, 16, 32);
#pragma unroll
        for (int r = 0; r < 8; ++r) rowx2[p][r] = __shfl(x2full, r + 8 * half, 32);
    }

    float minv[2][8];
    int   mini[2][8];
#pragma unroll
    for (int p = 0; p < 2; ++p)
#pragma unroll
        for (int r = 0; r < 8; ++r) { minv[p][r] = 3.4e38f; mini[p][r] = 0; }

    for (int ct = 0; ct < NT; ++ct) {
        __syncthreads();                       // staging of tile ct complete
        if (ct + 1 < NT) stage(ct + 1);        // overlap next-tile staging with compute
        const __bf16* tH = tileH[ct & 1];
        const __bf16* tL = tileL[ct & 1];

        float c2v = c2s[ct * 16 + lm];
        v8f acc[2][3] = {};                    // 6 independent chains
#pragma unroll
        for (int t = 0; t < 4; ++t) {
            const __bf16* rH = &tH[lm * STRH + 32 * t + 16 * half];
            const __bf16* rL = &tL[lm * STRH + 32 * t + 16 * half];
            BPair ph = { *(const v8bf*)rH, *(const v8bf*)(rH + 8) };
            BPair pl = { *(const v8bf*)rL, *(const v8bf*)(rL + 8) };
            v16bf bh = __builtin_bit_cast(v16bf, ph);
            v16bf bl = __builtin_bit_cast(v16bf, pl);
#pragma unroll
            for (int p = 0; p < 2; ++p) {
                acc[p][0] = __builtin_amdgcn_wmma_f32_16x16x32_bf16(
                    false, ah[p][t], false, bh, (short)0, acc[p][0], false, false);
                acc[p][1] = __builtin_amdgcn_wmma_f32_16x16x32_bf16(
                    false, ah[p][t], false, bl, (short)0, acc[p][1], false, false);
                acc[p][2] = __builtin_amdgcn_wmma_f32_16x16x32_bf16(
                    false, al[p][t], false, bh, (short)0, acc[p][2], false, false);
            }
        }
        // epilogue: lane holds (row r + 8*half, col lm) of each tile
        int col = ct * 16 + lm;
#pragma unroll
        for (int p = 0; p < 2; ++p)
#pragma unroll
            for (int r = 0; r < 8; ++r) {
                float dot  = acc[p][0][r] + (acc[p][1][r] + acc[p][2][r]);
                float dist = rowx2[p][r] + c2v - 2.0f * dot;
                if (dist < minv[p][r]) { minv[p][r] = dist; mini[p][r] = col; }
            }
    }

    // cross-lane argmin within each 16-lane group (first-min tie-break on index)
#pragma unroll
    for (int p = 0; p < 2; ++p)
#pragma unroll
        for (int r = 0; r < 8; ++r) {
            float v = minv[p][r]; int i = mini[p][r];
#pragma unroll
            for (int off = 8; off >= 1; off >>= 1) {
                float ov = __shfl_xor(v, off, 32);
                int   oi = __shfl_xor(i, off, 32);
                if (ov < v || (ov == v && oi < i)) { v = ov; i = oi; }
            }
            minv[p][r] = v; mini[p][r] = i;
        }

    if (lm == 0) {  // lanes 0 (rows 0..7) and 16 (rows 8..15) per tile
        float lsum = 0.0f;
#pragma unroll
        for (int p = 0; p < 2; ++p)
#pragma unroll
            for (int r = 0; r < 8; ++r) {
                int row = rowBase + p * 16 + r + 8 * half;
                labels[row]   = mini[p][r];
                labels_f[row] = (float)mini[p][r];
                lsum += minv[p][r];
            }
        atomicAdd(lossacc, lsum);
    }
}

// Segmented sums: block = (center-group, sample-chunk); LDS accumulators, global flush.
__global__ __launch_bounds__(256)
void k_scatter(const float* __restrict__ emb, const int* __restrict__ labels,
               float* __restrict__ sums, int* __restrict__ cnts) {
    __shared__ float lsum[CPG * DIM];  // 32 KB
    __shared__ int   lcnt[CPG];
    const int tid   = threadIdx.x;
    const int lane  = tid & 31;
    const int wave  = tid >> 5;
    const int g     = blockIdx.x % GROUPS;
    const int chunk = blockIdx.x / GROUPS;

    for (int j = tid; j < CPG * DIM; j += 256) lsum[j] = 0.0f;
    if (tid < CPG) lcnt[tid] = 0;
    __syncthreads();

    const int per = N_SAMPLES / CHUNKS;
    const int s0  = chunk * per;
    for (int s = s0 + wave; s < s0 + per; s += 8) {
        int c = labels[s];
        if ((c >> 6) == g) {
            int cl2 = c & (CPG - 1);
#pragma unroll
            for (int j = 0; j < 4; ++j) {
                int d = lane + 32 * j;
                atomicAdd(&lsum[cl2 * DIM + d], emb[s * DIM + d]);
            }
            if (lane == 0) atomicAdd(&lcnt[cl2], 1);
        }
    }
    __syncthreads();

    for (int j = tid; j < CPG * DIM; j += 256) {
        float v = lsum[j];
        if (v != 0.0f) atomicAdd(&sums[g * CPG * DIM + j], v);
    }
    if (tid < CPG) { int v = lcnt[tid]; if (v) atomicAdd(&cnts[g * CPG + tid], v); }
}

// Closed-form running-mean result + counts + loss.
__global__ void k_finalize(const float* __restrict__ centers, const int* __restrict__ count0,
                           const float* __restrict__ sums, const int* __restrict__ cnts,
                           const float* __restrict__ lossacc, float* __restrict__ dout) {
    int idx = blockIdx.x * 256 + threadIdx.x;
    if (idx < KCENT * DIM) {
        int c = idx / DIM;
        float n0    = (float)count0[c];
        float denom = n0 + (float)cnts[c];
        dout[1 + N_SAMPLES + idx] = (n0 * centers[idx] + sums[idx]) / denom;
    } else if (idx < KCENT * DIM + KCENT) {
        int c = idx - KCENT * DIM;
        dout[1 + N_SAMPLES + KCENT * DIM + c] = (float)(count0[c] + cnts[c]);
    } else if (idx == KCENT * DIM + KCENT) {
        dout[0] = *lossacc / (float)N_SAMPLES;
    }
}

extern "C" void kernel_launch(void* const* d_in, const int* in_sizes, int n_in,
                              void* d_out, int out_size, void* d_ws, size_t ws_size,
                              hipStream_t stream) {
    const float* emb     = (const float*)d_in[0];
    const float* centers = (const float*)d_in[1];
    const int*   count0  = (const int*)d_in[2];
    float* dout = (float*)d_out;

    float*  ws      = (float*)d_ws;
    float*  c2      = ws + WS_C2;
    float*  lossacc = ws + WS_LOSS;
    int*    labels  = (int*)(ws + WS_LABELS);
    float*  sums    = ws + WS_SUMS;
    int*    cnts    = (int*)(ws + WS_CNT);
    __bf16* ch      = (__bf16*)(ws + WS_CH);
    __bf16* cl      = (__bf16*)(ws + WS_CL);

    k_initws<<<(KCENT * DIM + 255) / 256, 256, 0, stream>>>(sums, cnts, lossacc);
    k_prep<<<KCENT, DIM, 0, stream>>>(centers, c2, ch, cl);
    k_dist_argmin<<<N_SAMPLES / 128, 128, 0, stream>>>(emb, ch, cl, c2, labels,
                                                       dout + 1, lossacc);
    k_scatter<<<GROUPS * CHUNKS, 256, 0, stream>>>(emb, labels, sums, cnts);
    k_finalize<<<(KCENT * DIM + KCENT + 1 + 255) / 256, 256, 0, stream>>>(
        centers, count0, sums, cnts, lossacc, dout);
}
